// BeliefStateWrapper_22849226015448
// MI455X (gfx1250) — matrix operand order repaired and checked by reading, not compiled, not run on hardware
//
#include <hip/hip_runtime.h>
#include <hip/hip_bf16.h>

// ---------------------------------------------------------------------------
// BeliefStateWrapper fused pipeline for MI455X (gfx1250, wave32, WMMA).
// B=2, L=512, D=512, V=32000; N pairs from in_sizes[3].
//   1) convert/transpose w1, w2 to bf16 [N,K] row-major (one-time; w2^T bf16
//      = 65.5 MB -> resident in 192 MB L2 for the whole GEMM2 sweep)
//   2) gather fb embeddings -> bf16 A [Mp,1024], zero padded
//   3) GEMM1 (WMMA bf16 16x16x32, f32 acc) + bias + LeakyReLU -> bf16 H
//   4) GEMM2 fused with online logsumexp + label-logit capture; the block's
//      16-row H tile is staged in LDS once (A frags via ds_load_b128), and
//      each wave runs two independent WMMA accumulation chains per A frag
//   5) deterministic final reduction -> scalar mean
// ---------------------------------------------------------------------------

#define LSEQ 512
#define DIM  512
#define VOC  32000
#define K1   1024          // 2*DIM
#define NEG_BIG (-3.0e38f)

typedef __attribute__((ext_vector_type(16))) __bf16 v16bf;
typedef __attribute__((ext_vector_type(8)))  float  v8f;

union FragU { uint4 u[2]; v16bf v; };

__device__ __forceinline__ unsigned short f2bf(float f) {
    union { float f; unsigned u; } x; x.f = f;
    unsigned u = x.u;
    unsigned r = u + 0x7FFFu + ((u >> 16) & 1u);   // round-to-nearest-even
    return (unsigned short)(r >> 16);
}

// W [K,N] fp32 row-major  ->  WT [N,K] bf16 row-major
__global__ void convert_transpose(const float* __restrict__ W,
                                  unsigned short* __restrict__ WT,
                                  int K, int N) {
    long long idx = (long long)blockIdx.x * blockDim.x + threadIdx.x;
    long long total = (long long)K * N;
    if (idx >= total) return;
    int k = (int)(idx / N);
    int n = (int)(idx - (long long)k * N);        // coalesced read along n
    WT[(long long)n * K + k] = f2bf(W[idx]);
}

// Build A [Mp, 1024] bf16: row r = concat(fwd[b, fi[n], :], bwd[b, bi[n], :])
__global__ void gather_pack(const float* __restrict__ fwd,
                            const float* __restrict__ bwd,
                            const int* __restrict__ fi,
                            const int* __restrict__ bi,
                            unsigned short* __restrict__ A,
                            int Mreal, int Np) {
    int r = blockIdx.x;
    int t = threadIdx.x;                          // 256 threads
    unsigned short* row = A + (size_t)r * K1;
    if (r < Mreal) {
        int b = r / Np, n = r - b * Np;
        const float* s0 = fwd + ((size_t)b * LSEQ + fi[n]) * DIM;
        const float* s1 = bwd + ((size_t)b * LSEQ + bi[n]) * DIM;
        for (int e = t; e < DIM; e += 256) {
            row[e]       = f2bf(s0[e]);
            row[DIM + e] = f2bf(s1[e]);
        }
    } else {
        for (int e = t; e < K1; e += 256) row[e] = 0;
    }
}

// A frag: dense 16-bit A layout (lane = m, K pair-interleave halves)
__device__ __forceinline__ v16bf load_a_frag(const unsigned short* base,
                                             int lda, int m0, int k0, int lane) {
    FragU a;
    const uint4* p = (const uint4*)(base + (size_t)(m0 + (lane & 15)) * lda + k0);
    int sel = lane >> 4;
    a.u[0] = p[sel];        // K 0..7   (or 8..15 for upper half-wave)
    a.u[1] = p[2 + sel];    // K 16..23 (or 24..31)
    return a.v;
}

// B frag from [N,K] row-major: lane = n, K pair-interleave halves (same as A)
__device__ __forceinline__ v16bf load_b_frag(const unsigned short* base,
                                             int ldb, int n0, int k0, int lane) {
    FragU b;
    const uint4* p = (const uint4*)(base + (size_t)(n0 + (lane & 15)) * ldb + k0);
    int sel = lane >> 4;
    b.u[0] = p[sel];
    b.u[1] = p[2 + sel];
    return b.v;
}

// H = leaky_relu(A @ W1T^T + b1), bf16 out. Block: 256 thr = 8 waves,
// wave tile grid 2x4 -> 32(M) x 64(N) per workgroup.
__global__ void gemm1(const unsigned short* __restrict__ A,
                      const unsigned short* __restrict__ W1T,
                      const float* __restrict__ b1,
                      unsigned short* __restrict__ H) {
    int tid  = threadIdx.x;
    int wave = tid >> 5, lane = tid & 31;
    int tm = blockIdx.x * 32 + (wave >> 2) * 16;
    int tn = blockIdx.y * 64 + (wave & 3) * 16;
    v8f acc = {};
    for (int k0 = 0; k0 < K1; k0 += 32) {
        v16bf af = load_a_frag(A,   K1, tm, k0, lane);
        v16bf bf = load_b_frag(W1T, K1, tn, k0, lane);
        acc = __builtin_amdgcn_wmma_f32_16x16x32_bf16(
                  false, af, false, bf, (short)0, acc, false, false);
    }
    int col   = tn + (lane & 15);
    float bia = b1[col];
    int rbase = tm + ((lane >= 16) ? 8 : 0);
#pragma unroll
    for (int v = 0; v < 8; v++) {
        float x = acc[v] + bia;
        x = (x > 0.0f) ? x : 0.01f * x;           // LeakyReLU(0.01)
        H[(size_t)(rbase + v) * DIM + col] = f2bf(x);
    }
}

// Fused GEMM2 + online logsumexp + label logit. Block = (16-row tile, branch).
// A tile (16x512 bf16 = 16 KB) staged in LDS once; each wave sweeps pairs of
// column tiles -> two independent WMMA chains per A fragment.
__global__ void gemm2_lse(const unsigned short* __restrict__ H,
                          const unsigned short* __restrict__ W2T,
                          const float* __restrict__ b2,
                          const int* __restrict__ seq,
                          const int* __restrict__ fi,
                          const int* __restrict__ bi,
                          float* __restrict__ partials,
                          int Mreal, int Np) {
    int rb = blockIdx.x, branch = blockIdx.y;
    int tid  = threadIdx.x;
    int wave = tid >> 5, lane = tid & 31;
    int m0 = rb * 16;

    __shared__ unsigned short Atile[16 * DIM];    // 16 KB
    __shared__ float2 red[8][32][8];              // 16 KB
    __shared__ float  labLDS[16];
    __shared__ float  nllws[16];
    if (tid < 16) labLDS[tid] = 0.0f;

    // cooperative stage of the block's 16 H rows into LDS (1024 uint4 total)
    {
        const uint4* src = (const uint4*)(H + (size_t)m0 * DIM);
        uint4*       dst = (uint4*)Atile;
#pragma unroll
        for (int i = 0; i < 4; i++) dst[tid + 256 * i] = src[tid + 256 * i];
    }
    __syncthreads();

    float rm[8], rs[8];
    int labels[8];
#pragma unroll
    for (int v = 0; v < 8; v++) {
        rm[v] = NEG_BIG; rs[v] = 0.0f;
        int r = m0 + v + ((lane >= 16) ? 8 : 0);
        int lab = -1;
        if (r < Mreal) {
            int b = r / Np, n = r - b * Np;
            int pos = (branch == 0) ? fi[n] : bi[n];
            lab = seq[b * LSEQ + pos];
        }
        labels[v] = lab;
    }

    const int npairs = VOC / 32;                  // 1000 tile-pairs, uniform
    const unsigned short* Wb = W2T + (size_t)branch * VOC * DIM;
    for (int tp = wave; tp < npairs; tp += 8) {
        int c0 = tp * 32;
        if (tp + 8 < npairs)                      // pull next B pair toward L0/L2
            __builtin_prefetch(Wb + (size_t)((tp + 8) * 32 + (lane & 15)) * DIM, 0, 1);
        v8f acc0 = {}, acc1 = {};
        for (int k0 = 0; k0 < DIM; k0 += 32) {
            v16bf af  = load_a_frag(Atile, DIM, 0, k0, lane);   // LDS
            v16bf bf0 = load_b_frag(Wb, DIM, c0,      k0, lane);
            v16bf bf1 = load_b_frag(Wb, DIM, c0 + 16, k0, lane);
            acc0 = __builtin_amdgcn_wmma_f32_16x16x32_bf16(
                       false, af, false, bf0, (short)0, acc0, false, false);
            acc1 = __builtin_amdgcn_wmma_f32_16x16x32_bf16(
                       false, af, false, bf1, (short)0, acc1, false, false);
        }
#pragma unroll
        for (int half = 0; half < 2; half++) {
            int col   = c0 + half * 16 + (lane & 15);
            float bia = b2[branch * VOC + col];
#pragma unroll
            for (int v = 0; v < 8; v++) {
                float x = (half ? acc1[v] : acc0[v]) + bia;
                if (labels[v] == col) labLDS[v + ((lane >= 16) ? 8 : 0)] = x;
                float nm = fmaxf(rm[v], x);
                rs[v] = rs[v] * expf(rm[v] - nm) + expf(x - nm);
                rm[v] = nm;
            }
        }
    }
#pragma unroll
    for (int v = 0; v < 8; v++) red[wave][lane][v] = make_float2(rm[v], rs[v]);
    __syncthreads();

    if (tid < 16) {                               // fixed-order combine: deterministic
        int row = tid;
        int lo  = (row < 8) ? 0 : 16;
        int v   = row & 7;
        float m = NEG_BIG, s = 0.0f;
        for (int w = 0; w < 8; w++)
            for (int l = lo; l < lo + 16; l++) {
                float2 p = red[w][l][v];
                float nm = fmaxf(m, p.x);
                s = s * expf(m - nm) + p.y * expf(p.x - nm);
                m = nm;
            }
        float lse = m + logf(s);
        int r = m0 + row;
        float nw = 0.0f;
        if (r < Mreal)
            nw = (lse - labLDS[row]) * ((branch == 0) ? 1.0f : 0.25f);
        nllws[row] = nw;
    }
    __syncthreads();
    if (tid == 0) {
        float s = 0.0f;
        for (int i = 0; i < 16; i++) s += nllws[i];
        partials[rb * 2 + branch] = s;
    }
}

__global__ void final_reduce(const float* __restrict__ partials, int n,
                             float* __restrict__ out, float invden) {
    if (blockIdx.x == 0 && threadIdx.x == 0) {
        float s = 0.0f;
        for (int i = 0; i < n; i++) s += partials[i];  // fixed order
        out[0] = s * invden;
    }
}

extern "C" void kernel_launch(void* const* d_in, const int* in_sizes, int n_in,
                              void* d_out, int out_size, void* d_ws, size_t ws_size,
                              hipStream_t stream) {
    (void)n_in; (void)out_size; (void)ws_size;
    const float* fwd = (const float*)d_in[0];
    const float* bwd = (const float*)d_in[1];
    const int*   seq = (const int*)d_in[2];
    const int*   fi  = (const int*)d_in[3];
    const int*   bi  = (const int*)d_in[4];
    const float* w1  = (const float*)d_in[5];
    const float* b1  = (const float*)d_in[6];
    const float* w2  = (const float*)d_in[7];
    const float* b2  = (const float*)d_in[8];

    int Np    = in_sizes[3];
    int Mreal = 2 * Np;
    int Mp    = ((Mreal + 63) / 64) * 64;         // pad for 32/16-row tiling

    char* ws = (char*)d_ws;
    unsigned short* w2t = (unsigned short*)ws;            size_t off = (size_t)VOC * 2 * DIM * 2;
    unsigned short* w1t = (unsigned short*)(ws + off);    off += (size_t)DIM * K1 * 2;
    unsigned short* Ab  = (unsigned short*)(ws + off);    off += (size_t)Mp * K1 * 2;
    unsigned short* Hb  = (unsigned short*)(ws + off);    off += (size_t)Mp * DIM * 2;
    float*          par = (float*)(ws + off);

    long long w1e = (long long)K1 * DIM;                  // 524288
    long long w2e = (long long)DIM * (2 * VOC);           // 32768000
    convert_transpose<<<dim3((unsigned)((w1e + 255) / 256)), 256, 0, stream>>>(w1, w1t, K1, DIM);
    convert_transpose<<<dim3((unsigned)((w2e + 255) / 256)), 256, 0, stream>>>(w2, w2t, DIM, 2 * VOC);
    gather_pack<<<dim3(Mp), 256, 0, stream>>>(fwd, bwd, fi, bi, Ab, Mreal, Np);
    gemm1<<<dim3(Mp / 32, DIM / 64), 256, 0, stream>>>(Ab, w1t, b1, Hb);
    gemm2_lse<<<dim3(Mp / 16, 2), 256, 0, stream>>>(Hb, w2t, b2, seq, fi, bi, par, Mreal, Np);
    final_reduce<<<1, 32, 0, stream>>>(par, (Mp / 16) * 2, (float*)d_out,
                                       1.0f / (float)(Mreal * 2));
}
